// gru_decoder_t_74071005987134
// MI455X (gfx1250) — matrix operand order repaired and checked by reading, not compiled, not run on hardware
//
#include <hip/hip_runtime.h>
#include <hip/hip_bf16.h>
#include <stdint.h>

#define B_  128
#define NH_ 512
#define NG_ 300
#define NW_ 30000
#define T_  20

typedef __attribute__((ext_vector_type(16))) __bf16 bf16x16;
typedef __attribute__((ext_vector_type(8)))  float  f32x8;
typedef unsigned short u16;
typedef unsigned int   u32;
typedef unsigned long long u64;

// fp32 -> bf16 round-to-nearest-even
__device__ __forceinline__ u16 f2bf(float f) {
  u32 u = __float_as_uint(f);
  u32 r = u + 0x7FFFu + ((u >> 16) & 1u);
  return (u16)(r >> 16);
}

// monotonic key for float compare via unsigned compare
__device__ __forceinline__ u32 fordkey(float f) {
  u32 u = __float_as_uint(f);
  return u ^ ((u32)((int)u >> 31) | 0x80000000u);
}

union FragU { uint4 q[2]; bf16x16 v; };

// A fragment: 16x32 bf16, M x K, row-major. ISA layout:
// lanes 0-15: row=lane, v0..3 = K 0..7, v4..7 = K 16..23
// lanes16-31: row=lane-16, v0..3 = K 8..15, v4..7 = K 24..31
__device__ __forceinline__ bf16x16 load_a(const u16* base, int ld, int row0, int k0, int lane) {
  int r = row0 + (lane & 15);
  int k = k0 + ((lane >> 4) << 3);
  const u16* p = base + (size_t)r * ld + k;
  FragU f;
  f.q[0] = *(const uint4*)(p);
  f.q[1] = *(const uint4*)(p + 16);
  return f.v;
}

// B fragment: 32x16 bf16 (K x N), sourced from weight W stored N x K row-major.
// ISA layout: lanes 0-15: col=lane, v0..3 = K 0..7, v4..7 = K 16..23; lanes 16-31: K 8..15, 24..31.
__device__ __forceinline__ bf16x16 load_b(const u16* base, int ld, int n0, int k0, int lane) {
  int n = n0 + (lane & 15);
  int k = k0 + ((lane >> 4) << 3);
  const u16* p = base + (size_t)n * ld + k;
  FragU f;
  f.q[0] = *(const uint4*)(p);
  f.q[1] = *(const uint4*)(p + 16);
  return f.v;
}

__device__ __forceinline__ u64 shfl_xor_u64_16(u64 v, int mask) {
  u32 lo = (u32)v, hi = (u32)(v >> 32);
  lo = __shfl_xor(lo, mask, 16);
  hi = __shfl_xor(hi, mask, 16);
  return ((u64)hi << 32) | lo;
}

// Generic GEMM: C[M=128, ncols] = act(A(MxK) @ W(ncols x K)^T + bias)
// 8 waves / block, each wave one 16x16 tile. grid.x covers N tiles, grid.y = 8 M tiles.
__global__ void __launch_bounds__(256)
gemm_bf16(const u16* __restrict__ A, int lda,
          const u16* __restrict__ W, int ldb,
          const float* __restrict__ bias,
          float* __restrict__ outF, int ldf,
          u16* __restrict__ outB, int ldob,
          int ktiles, int ntiles, int ncols, int act)
{
  int wave = threadIdx.x >> 5;
  int lane = threadIdx.x & 31;
  int nt = blockIdx.x * 8 + wave;
  if (nt >= ntiles) return;            // wave-uniform: EXEC stays all-1s for WMMA
  int mt = blockIdx.y;

  f32x8 acc = {};
  for (int kt = 0; kt < ktiles; ++kt) {
    bf16x16 a = load_a(A, lda, mt * 16, kt * 32, lane);
    bf16x16 b = load_b(W, ldb, nt * 16, kt * 32, lane);
    acc = __builtin_amdgcn_wmma_f32_16x16x32_bf16(false, a, false, b, (short)0, acc, false, false);
  }

  int n  = nt * 16 + (lane & 15);
  int mb = mt * 16 + ((lane >> 4) << 3);
  if (n >= ncols) return;
  float bv = bias ? bias[n] : 0.f;
#pragma unroll
  for (int r = 0; r < 8; ++r) {
    float v = acc[r] + bv;
    if (act) v = v > 0.f ? v : 0.01f * v;   // LeakyReLU(0.01)
    int m = mb + r;
    if (outF) outF[(size_t)m * ldf  + n] = v;
    if (outB) outB[(size_t)m * ldob + n] = f2bf(v);
  }
}

// Big vocab GEMM: wp = p(128x320bf16) @ Wp2(30000x320bf16)^T + bp2 -> preds, fused argmax.
// Each wave computes a 64x16 strip (4 M-tiles) so every B fragment is reused 4x from
// registers: Wp2 L2 traffic drops from 150 MB to 37.5 MB per step.
__global__ void __launch_bounds__(256)
gemm_wp_argmax(const u16* __restrict__ A, const u16* __restrict__ W,
               const float* __restrict__ bias,
               float* __restrict__ outF, u64* __restrict__ amax)
{
  __shared__ u64 smax[64];
  int wave = threadIdx.x >> 5;
  int lane = threadIdx.x & 31;
  if (threadIdx.x < 64) smax[threadIdx.x] = 0ull;
  __syncthreads();

  int nt = blockIdx.x * 8 + wave;
  int m0 = blockIdx.y * 64;                  // 4 M-tiles = 64 rows per block
  if (nt < (NW_ / 16)) {                     // wave-uniform guard: EXEC all-1s
    f32x8 acc[4] = {};
    for (int kt = 0; kt < 10; ++kt) {        // K = 320
      bf16x16 b = load_b(W, 320, nt * 16, kt * 32, lane);
#pragma unroll
      for (int mi = 0; mi < 4; ++mi) {
        bf16x16 a = load_a(A, 320, m0 + mi * 16, kt * 32, lane);
        acc[mi] = __builtin_amdgcn_wmma_f32_16x16x32_bf16(false, a, false, b,
                                                          (short)0, acc[mi], false, false);
      }
    }
    int n = nt * 16 + (lane & 15);
    float bv = bias[n];
#pragma unroll
    for (int mi = 0; mi < 4; ++mi) {
      int mb = m0 + mi * 16 + ((lane >> 4) << 3);
#pragma unroll
      for (int r = 0; r < 8; ++r) {
        float v = acc[mi][r] + bv;
        outF[(size_t)(mb + r) * NW_ + n] = v;
        // pack: high = ordered float key, low = ~n so ties pick smallest index
        u64 pk = ((u64)fordkey(v) << 32) | (u32)(~(u32)n);
#pragma unroll
        for (int off = 8; off > 0; off >>= 1) {
          u64 o = shfl_xor_u64_16(pk, off);
          if (o > pk) pk = o;
        }
        if ((lane & 15) == r)
          atomicMax(&smax[mi * 16 + ((lane >> 4) << 3) + r], pk);
      }
    }
  }
  __syncthreads();
  if (threadIdx.x < 64)
    atomicMax(&amax[m0 + threadIdx.x], smax[threadIdx.x]);
}

// GRU elementwise: h_new from gi/gh; also zeroes amax for this step's argmax.
__global__ void gru_hnew(const float* __restrict__ gi, const float* __restrict__ gh,
                         float* __restrict__ h, u16* __restrict__ hbf,
                         float* __restrict__ hs_out, u64* __restrict__ amax)
{
  int i = blockIdx.x * blockDim.x + threadIdx.x;
  if (i < B_) amax[i] = 0ull;
  if (i >= B_ * NG_) return;
  int b = i / NG_, j = i - b * NG_;
  float ir = gi[b * 900 + j], iz = gi[b * 900 + 300 + j], in_ = gi[b * 900 + 600 + j];
  float hr = gh[b * 900 + j], hz = gh[b * 900 + 300 + j], hn  = gh[b * 900 + 600 + j];
  float r  = 1.f / (1.f + expf(-(ir + hr)));
  float z  = 1.f / (1.f + expf(-(iz + hz)));
  float nn = tanhf(in_ + r * hn);
  float hv = (1.f - z) * nn + z * h[i];
  h[i] = hv;
  hbf[b * 320 + j] = f2bf(hv);
  hs_out[i] = hv;
}

// argmax finalize + embedding gather into xin cols [512, 812) (bf16)
__global__ void gather_emb(const u64* __restrict__ amax, const float* __restrict__ embt,
                           u16* __restrict__ xin)
{
  int b = blockIdx.x;
  u32 n = ~(u32)(amax[b]);               // recover argmax index
  const float* row = embt + (size_t)(n + 1) * NG_;
  for (int j = threadIdx.x; j < NG_; j += blockDim.x)
    xin[b * 832 + 512 + j] = f2bf(row[j]);
}

// pad-convert fp32 (rows x cols) into bf16 (.. x ldd) buffer, zero padding
__global__ void cvt_pad(const float* __restrict__ src, u16* __restrict__ dst,
                        int rows, int cols, int ldd, int total)
{
  for (int i = blockIdx.x * blockDim.x + threadIdx.x; i < total; i += gridDim.x * blockDim.x) {
    int r = i / ldd, c = i - r * ldd;
    dst[i] = (r < rows && c < cols) ? f2bf(src[r * cols + c]) : (u16)0;
  }
}

__global__ void zf32(float* __restrict__ p, int n) {
  int i = blockIdx.x * blockDim.x + threadIdx.x;
  if (i < n) p[i] = 0.f;
}

extern "C" void kernel_launch(void* const* d_in, const int* in_sizes, int n_in,
                              void* d_out, int out_size, void* d_ws, size_t ws_size,
                              hipStream_t stream)
{
  const float* y    = (const float*)d_in[0];
  const float* embt = (const float*)d_in[1];
  const float* W1   = (const float*)d_in[2];
  const float* b1   = (const float*)d_in[3];
  const float* W2   = (const float*)d_in[4];
  const float* b2   = (const float*)d_in[5];
  const float* Wih  = (const float*)d_in[6];
  const float* Whh  = (const float*)d_in[7];
  const float* bih  = (const float*)d_in[8];
  const float* bhh  = (const float*)d_in[9];
  const float* Wp1  = (const float*)d_in[10];
  const float* bp1  = (const float*)d_in[11];
  const float* Wp2  = (const float*)d_in[12];
  const float* bp2  = (const float*)d_in[13];

  float* preds = (float*)d_out;
  float* hsout = preds + (size_t)T_ * B_ * NW_;

  char* ws = (char*)d_ws;
  size_t off = 0;
  auto carve = [&](size_t bytes) -> char* {
    char* p = ws + off;
    off = (off + bytes + 255) & ~(size_t)255;
    return p;
  };
  u16* Wp2b = (u16*)carve((size_t)NW_ * 320 * 2);  // 30000 x 320
  u16* Wihb = (u16*)carve((size_t)912 * 512 * 2);  // 900->912 rows, K=512
  u16* Whhb = (u16*)carve((size_t)912 * 320 * 2);  // K 300->320
  u16* Wp1b = (u16*)carve((size_t)304 * 320 * 2);
  u16* W1b  = (u16*)carve((size_t)512 * 832 * 2);  // K 812->832
  u16* W2b  = (u16*)carve((size_t)512 * 512 * 2);
  u16* xin  = (u16*)carve((size_t)B_ * 832 * 2);   // [y | emb | pad]
  u16* t1b  = (u16*)carve((size_t)B_ * 512 * 2);
  u16* cib  = (u16*)carve((size_t)B_ * 512 * 2);
  u16* hbf  = (u16*)carve((size_t)B_ * 320 * 2);
  u16* pbf  = (u16*)carve((size_t)B_ * 320 * 2);
  float* h  = (float*)carve((size_t)B_ * 300 * 4);
  float* gi = (float*)carve((size_t)B_ * 900 * 4);
  float* gh = (float*)carve((size_t)B_ * 900 * 4);
  u64* amax = (u64*)carve((size_t)B_ * 8);

  auto cvt = [&](const float* s, u16* d, int rows, int cols, int ldd, int rowsP) {
    int total = rowsP * ldd;
    int blocks = (total + 255) / 256;
    cvt_pad<<<blocks, 256, 0, stream>>>(s, d, rows, cols, ldd, total);
  };

  // --- one-time per launch: weight conversion + state init ---
  cvt(Wp2, Wp2b, NW_, NG_, 320, NW_);
  cvt(Wih, Wihb, 900, 512, 512, 912);
  cvt(Whh, Whhb, 900, 300, 320, 912);
  cvt(Wp1, Wp1b, 300, 300, 320, 304);
  cvt(W1,  W1b,  512, 812, 832, 512);
  cvt(W2,  W2b,  512, 512, 512, 512);
  cvt(y,   xin,  B_, 512, 832, B_);       // emb region + K-pad start at 0
  cvt(y,   hbf,  0, 0, 320, B_);          // zero (incl. K-pad)
  cvt(y,   pbf,  0, 0, 320, B_);          // zero (incl. K-pad)
  zf32<<<(B_ * NG_ + 255) / 256, 256, 0, stream>>>(h, B_ * NG_);

  // ci0 = hidden_map(concat(y, 0))
  gemm_bf16<<<dim3(4, 8), 256, 0, stream>>>(xin, 832, W1b, 832, b1,
                                            nullptr, 0, t1b, 512, 26, 32, 512, 1);
  gemm_bf16<<<dim3(4, 8), 256, 0, stream>>>(t1b, 512, W2b, 512, b2,
                                            nullptr, 0, cib, 512, 16, 32, 512, 0);

  for (int t = 0; t < T_; ++t) {
    // gi = ci @ W_ih^T + b_ih ; gh = h @ W_hh^T + b_hh   (N=900 -> 57 tiles)
    gemm_bf16<<<dim3(8, 8), 256, 0, stream>>>(cib, 512, Wihb, 512, bih,
                                              gi, 900, nullptr, 0, 16, 57, 900, 0);
    gemm_bf16<<<dim3(8, 8), 256, 0, stream>>>(hbf, 320, Whhb, 320, bhh,
                                              gh, 900, nullptr, 0, 10, 57, 900, 0);
    // h_new (also zeroes amax for this step)
    gru_hnew<<<(B_ * NG_ + 255) / 256, 256, 0, stream>>>(
        gi, gh, h, hbf, hsout + (size_t)t * B_ * NG_, amax);
    // p = lrelu(h_new @ Wp1^T + bp1)   (N=300 -> 19 tiles)
    gemm_bf16<<<dim3(3, 8), 256, 0, stream>>>(hbf, 320, Wp1b, 320, bp1,
                                              nullptr, 0, pbf, 320, 10, 19, 300, 1);
    // wp = p @ Wp2^T + bp2 -> preds[t], fused per-row argmax
    // 1875 N tiles, grid.y=2 x 64-row strips
    gemm_wp_argmax<<<dim3(235, 2), 256, 0, stream>>>(
        pbf, Wp2b, bp2, preds + (size_t)t * B_ * NW_, amax);
    // idx -> embedding -> xin cols [512,812)
    gather_emb<<<B_, 128, 0, stream>>>(amax, embt, xin);
    // ci_next = hidden_map(concat(y, emb))
    gemm_bf16<<<dim3(4, 8), 256, 0, stream>>>(xin, 832, W1b, 832, b1,
                                              nullptr, 0, t1b, 512, 26, 32, 512, 1);
    gemm_bf16<<<dim3(4, 8), 256, 0, stream>>>(t1b, 512, W2b, 512, b2,
                                              nullptr, 0, cib, 512, 16, 32, 512, 0);
  }
}
